// AttentionRef_15135464751327
// MI455X (gfx1250) — compile-verified, matrix operand order checked by reading, not run on hardware
//
#include <hip/hip_runtime.h>

// ---------------------------------------------------------------------------
// Fused attention block for MI455X (gfx1250, wave32, WMMA 16x16x32 f16).
//   0) pre-cast x, Wqkv, Wo -> f16 (once; GEMMs stream pure f16 operands)
//   1) qkv = x @ Wqkv^T + bqkv              (f16 WMMA, fp32 accum, 16x64/wave)
//   2) q,k = rmsnorm(q|k)*w ; v -> f16 transposed [h][d][token]
//   3) z   = softmax(q k^T) v   flash-style (f16 WMMA; score-transpose softmax,
//                                            denominator via ones-matrix WMMA)
//   4) out = z @ Wo^T + bo                  (f16 WMMA, fp32 accum/out)
// ---------------------------------------------------------------------------

#define SEQ    4096
#define DMODEL 1024
#define NH     16
#define DH     64

typedef _Float16 h16;
typedef __attribute__((ext_vector_type(16))) _Float16 v16h;
typedef __attribute__((ext_vector_type(8)))  float    v8f;

// ---------------------------------------------------------------------------
// Elementwise fp32 -> f16 cast (grid-stride).
// ---------------------------------------------------------------------------
__global__ void cast_f16_kernel(const float* __restrict__ src, h16* __restrict__ dst,
                                int n) {
  for (int i = blockIdx.x * blockDim.x + threadIdx.x; i < n; i += gridDim.x * blockDim.x)
    dst[i] = (h16)src[i];
}

// ---------------------------------------------------------------------------
// GEMM: C[M,N] = A[M,K] (f16 row-major) * B[N,K]^T (f16 row-major) + bias[N]
// Block = 128 threads (4 waves). Each wave computes a 16(M) x 64(N) C strip:
// one A operand is reused across 4 WMMAs per 32-wide K step.
// Operand layout (16-bit A/B): lane&15 = M (resp. N); element j<8 -> K =
// j + 8*kh, j>=8 -> K = 16 + (j-8) + 8*kh, kh = lane>>4 (contiguous 16B runs).
// C/D layout: N = lane&15, M = r + 8*kh for VGPR r.
// ---------------------------------------------------------------------------
__global__ void gemm_h16_kernel(const h16* __restrict__ A, const h16* __restrict__ B,
                                const float* __restrict__ bias, float* __restrict__ C,
                                int M, int N, int K) {
  const int lane = threadIdx.x & 31;
  const int wv   = threadIdx.x >> 5;
  const int nn   = lane & 15;
  const int kh   = lane >> 4;
  const int n0   = blockIdx.x * 64;
  const int m0   = (blockIdx.y * 4 + wv) * 16;

  const h16* arow = A + (size_t)(m0 + nn) * K;
  const h16* brow[4];
#pragma unroll
  for (int t = 0; t < 4; ++t) brow[t] = B + (size_t)(n0 + t * 16 + nn) * K;

  v8f acc[4] = {};
  for (int k0 = 0; k0 < K; k0 += 32) {
    __builtin_prefetch(brow[0] + k0 + 512, 0, 0);  // stream weights (global_prefetch)
    __builtin_prefetch(brow[2] + k0 + 512, 0, 0);

    const h16* pa = arow + k0 + 8 * kh;
    v16h a;
#pragma unroll
    for (int j = 0; j < 8; ++j) {
      a[j]     = pa[j];
      a[8 + j] = pa[16 + j];
    }
#pragma unroll
    for (int t = 0; t < 4; ++t) {
      const h16* pb = brow[t] + k0 + 8 * kh;
      v16h b;
#pragma unroll
      for (int j = 0; j < 8; ++j) {
        b[j]     = pb[j];
        b[8 + j] = pb[16 + j];
      }
      acc[t] = __builtin_amdgcn_wmma_f32_16x16x32_f16(false, a, false, b,
                                                      (short)0, acc[t], false, false);
    }
  }

#pragma unroll
  for (int t = 0; t < 4; ++t) {
    const int   col = n0 + t * 16 + nn;
    const float bv  = bias[col];
#pragma unroll
    for (int r = 0; r < 8; ++r) {
      C[(size_t)(m0 + r + 8 * kh) * N + col] = acc[t][r] + bv;
    }
  }
}

// ---------------------------------------------------------------------------
// RMSNorm(q), RMSNorm(k) -> f16 [h][token][64]; cast(v) -> f16 TRANSPOSED
// [h][d][token] so PV B-operand loads are contiguous along keys.
// One wave per (token, head); 2 elements per lane; shfl-xor reduction.
// ---------------------------------------------------------------------------
__global__ void rmsnorm_cast_kernel(const float* __restrict__ qkv,
                                    const float* __restrict__ wq,
                                    const float* __restrict__ wk,
                                    h16* __restrict__ Qh, h16* __restrict__ Kh,
                                    h16* __restrict__ Vt) {
  const int wid   = blockIdx.x * (blockDim.x >> 5) + (threadIdx.x >> 5);
  const int lane  = threadIdx.x & 31;
  const int token = wid >> 4;   // / NH
  const int h     = wid & 15;   // % NH

  const float* qrow = qkv + (size_t)token * (3 * DMODEL) + h * DH;
  const float* krow = qrow + DMODEL;
  const float* vrow = qrow + 2 * DMODEL;

  const float q0 = qrow[lane], q1 = qrow[lane + 32];
  const float k0 = krow[lane], k1 = krow[lane + 32];
  float sq = q0 * q0 + q1 * q1;
  float sk = k0 * k0 + k1 * k1;
#pragma unroll
  for (int off = 16; off > 0; off >>= 1) {
    sq += __shfl_xor(sq, off, 32);
    sk += __shfl_xor(sk, off, 32);
  }
  const float rq = rsqrtf(sq * (1.0f / DH) + 1e-6f);
  const float rk = rsqrtf(sk * (1.0f / DH) + 1e-6f);

  const size_t base = ((size_t)h * SEQ + token) * DH;
  Qh[base + lane]      = (h16)(q0 * rq * wq[lane]);
  Qh[base + lane + 32] = (h16)(q1 * rq * wq[lane + 32]);
  Kh[base + lane]      = (h16)(k0 * rk * wk[lane]);
  Kh[base + lane + 32] = (h16)(k1 * rk * wk[lane + 32]);
  // transposed V: Vt[h][d][token]
  Vt[((size_t)h * DH + lane)      * SEQ + token] = (h16)vrow[lane];
  Vt[((size_t)h * DH + lane + 32) * SEQ + token] = (h16)vrow[lane + 32];
}

// ---------------------------------------------------------------------------
// Flash attention. Block = 128 threads (4 waves); each wave owns one 16-query
// tile and a private LDS slice. KV block = 32 keys.
//  - S tile: 2 WMMAs per 16-key subtile (K-dim 64 = 2x32), fp32 accum.
//  - Softmax via score transpose: raw S goes through LDS once; each lane then
//    holds the 16 scores of its own A-operand slot (row nn, its key-half), so
//    the row max is 15 local VALU maxes + ONE shfl_xor(16) to merge halves,
//    and exp(s-m) casts directly into the A operand (no second LDS trip).
//  - alpha (per-row, needed in C layout) published via tiny LDS broadcast.
//  - Row sum: extra WMMA against an all-ones B rides as a 5th accumulator.
// No 1/sqrt(d) scaling, no causal mask (matches reference).
// ---------------------------------------------------------------------------
__global__ void flash_attn_kernel(const h16* __restrict__ Qh, const h16* __restrict__ Kh,
                                  const h16* __restrict__ Vt, h16* __restrict__ Z) {
  __shared__ float Slds[4][16 * 32];
  __shared__ float Alds[4][16];

  const int lane = threadIdx.x & 31;
  const int wv   = threadIdx.x >> 5;
  const int nn   = lane & 15;
  const int kh   = lane >> 4;
  const int h    = blockIdx.y;
  const int q0   = (blockIdx.x * 4 + wv) * 16;

  const h16* Qb = Qh + ((size_t)h * SEQ + q0) * DH;
  const h16* Kb = Kh + (size_t)h * SEQ * DH;
  const h16* Vb = Vt + (size_t)h * DH * SEQ;

  // Q tile as two A operands covering K-dim [0,32) and [32,64).
  v16h aq[2];
#pragma unroll
  for (int c = 0; c < 2; ++c) {
    const h16* p = Qb + (size_t)nn * DH + c * 32 + 8 * kh;
#pragma unroll
    for (int j = 0; j < 8; ++j) {
      aq[c][j]     = p[j];
      aq[c][8 + j] = p[16 + j];
    }
  }

  v16h bones;
#pragma unroll
  for (int j = 0; j < 16; ++j) bones[j] = (h16)1.0f;

  v8f   o[4] = {};         // O tile: 16 q-rows x 64 dims (4 N-tiles)
  v8f   lacc = {};         // softmax denominator (rowsum accumulator)
  float mrun = -1e30f;     // running max of row nn (replicated in both halves)

  for (int kt = 0; kt < SEQ; kt += 32) {
    __builtin_prefetch(Kb + (size_t)(kt + 32) * DH, 0, 0);
    __builtin_prefetch(Vb + kt + 32, 0, 0);

    // ---- scores S[16q x 32keys] as two 16x16 tiles ----
    v8f s[2] = {};
#pragma unroll
    for (int t = 0; t < 2; ++t) {
#pragma unroll
      for (int c = 0; c < 2; ++c) {
        // B = K^T (64x16): B[kk][n] = Kh[key kt+t*16+n][kk]
        const h16* p = Kb + (size_t)(kt + t * 16 + nn) * DH + c * 32 + 8 * kh;
        v16h b;
#pragma unroll
        for (int j = 0; j < 8; ++j) {
          b[j]     = p[j];
          b[8 + j] = p[16 + j];
        }
        s[t] = __builtin_amdgcn_wmma_f32_16x16x32_f16(false, aq[c], false, b,
                                                      (short)0, s[t], false, false);
      }
    }

    // ---- transpose raw scores through LDS: [m][key] fp32 ----
#pragma unroll
    for (int t = 0; t < 2; ++t)
#pragma unroll
      for (int r = 0; r < 8; ++r)
        Slds[wv][(r + 8 * kh) * 32 + t * 16 + nn] = s[t][r];
    __syncthreads();

    // ---- each lane: the 16 scores of its A-operand slot (row nn, half kh) ----
    float sv[16];
    {
      const float* sp = &Slds[wv][nn * 32 + 8 * kh];
#pragma unroll
      for (int j = 0; j < 8; ++j) {
        sv[j]     = sp[j];
        sv[8 + j] = sp[16 + j];
      }
    }
    float mt = sv[0];
#pragma unroll
    for (int j = 1; j < 16; ++j) mt = fmaxf(mt, sv[j]);
    mt = fmaxf(mt, __shfl_xor(mt, 16, 32));   // merge the two key-halves

    const float mnew    = fmaxf(mrun, mt);
    const float myalpha = __expf(mrun - mnew);
    mrun = mnew;

    // exp -> A operand directly (lane-local)
    v16h ap;
#pragma unroll
    for (int j = 0; j < 16; ++j) ap[j] = (h16)__expf(sv[j] - mnew);

    // publish alpha per row (lanes nn and nn+16 write identical values)
    Alds[wv][nn] = myalpha;
    __syncthreads();   // also fences this iteration's Slds reads vs next write

    float alpha[8];
    {
      const float* apv = &Alds[wv][8 * kh];
#pragma unroll
      for (int r = 0; r < 8; ++r) alpha[r] = apv[r];
    }

    // ---- rescale O and denominator, then O += P @ V ; l += P @ 1 ----
#pragma unroll
    for (int t = 0; t < 4; ++t)
#pragma unroll
      for (int r = 0; r < 8; ++r) o[t][r] *= alpha[r];
#pragma unroll
    for (int r = 0; r < 8; ++r) lacc[r] *= alpha[r];

#pragma unroll
    for (int t = 0; t < 4; ++t) {
      // B = V (32keys x 16dims): B[kk][n] = Vt[h][t*16+n][kt+kk] (contiguous runs)
      const h16* p = Vb + (size_t)(t * 16 + nn) * SEQ + kt + 8 * kh;
      v16h bv;
#pragma unroll
      for (int j = 0; j < 8; ++j) {
        bv[j]     = p[j];
        bv[8 + j] = p[16 + j];
      }
      o[t] = __builtin_amdgcn_wmma_f32_16x16x32_f16(false, ap, false, bv,
                                                    (short)0, o[t], false, false);
    }
    lacc = __builtin_amdgcn_wmma_f32_16x16x32_f16(false, ap, false, bones,
                                                  (short)0, lacc, false, false);
  }

  // ---- normalize and store Z[token][h*64 + d] (f16, token-major) ----
#pragma unroll
  for (int r = 0; r < 8; ++r) {
    const int   token = q0 + r + 8 * kh;
    const float inv   = 1.0f / lacc[r];
#pragma unroll
    for (int t = 0; t < 4; ++t) {
      Z[(size_t)token * DMODEL + h * DH + t * 16 + nn] = (h16)(o[t][r] * inv);
    }
  }
}

// ---------------------------------------------------------------------------
// Host-side launcher.
// Inputs: x[1,4096,1024] f32, Wqkv[3072,1024] f32, bqkv[3072] f32,
//         Wo[1024,1024] f32, bo[1024] f32, wq[64] f32, wk[64] f32.
// Output: f32 [4096*1024].
// Workspace layout (bytes):
//   [0,        48M)   qkv fp32  (4096 x 3072)
//   [+48M,     +56M)  Qh    f16 [h][token][64]
//   [+56M,     +64M)  Kh    f16 [h][token][64]
//   [+64M,     +72M)  Vt    f16 [h][d][token]
//   [+72M,     +80M)  Zh    f16 [token][1024]
//   [+80M,     +88M)  xh    f16 (4096 x 1024)
//   [+88M,     +94M)  Wqkvh f16 (3072 x 1024)
//   [+94M,     +96M)  Woh   f16 (1024 x 1024)
// ---------------------------------------------------------------------------
extern "C" void kernel_launch(void* const* d_in, const int* in_sizes, int n_in,
                              void* d_out, int out_size, void* d_ws, size_t ws_size,
                              hipStream_t stream) {
  const float* x    = (const float*)d_in[0];
  const float* Wqkv = (const float*)d_in[1];
  const float* bqkv = (const float*)d_in[2];
  const float* Wo   = (const float*)d_in[3];
  const float* bo   = (const float*)d_in[4];
  const float* wq   = (const float*)d_in[5];
  const float* wk   = (const float*)d_in[6];

  char*  ws    = (char*)d_ws;
  float* qkv   = (float*)ws;
  h16*   Qh    = (h16*)(ws + 50331648);
  h16*   Kh    = (h16*)(ws + 58720256);
  h16*   Vt    = (h16*)(ws + 67108864);
  h16*   Zh    = (h16*)(ws + 75497472);
  h16*   xh    = (h16*)(ws + 83886080);
  h16*   Wqkvh = (h16*)(ws + 92274688);
  h16*   Woh   = (h16*)(ws + 98566144);
  float* out   = (float*)d_out;

  // 0) one-time f16 casts of activations and weights
  cast_f16_kernel<<<2048, 256, 0, stream>>>(x, xh, SEQ * DMODEL);
  cast_f16_kernel<<<2048, 256, 0, stream>>>(Wqkv, Wqkvh, 3 * DMODEL * DMODEL);
  cast_f16_kernel<<<2048, 256, 0, stream>>>(Wo, Woh, DMODEL * DMODEL);

  // 1) QKV projection: [4096,1024] x [3072,1024]^T  (16x64 per wave)
  gemm_h16_kernel<<<dim3((3 * DMODEL) / 64, SEQ / 64), 128, 0, stream>>>(
      xh, Wqkvh, bqkv, qkv, SEQ, 3 * DMODEL, DMODEL);

  // 2) RMSNorm q,k + cast/transpose v
  rmsnorm_cast_kernel<<<(SEQ * NH) / 4, 128, 0, stream>>>(qkv, wq, wk, Qh, Kh, Vt);

  // 3) Flash attention; 4 q-tiles per block
  flash_attn_kernel<<<dim3(SEQ / 64, NH), 128, 0, stream>>>(Qh, Kh, Vt, Zh);

  // 4) Output projection: [4096,1024] x [1024,1024]^T
  gemm_h16_kernel<<<dim3(DMODEL / 64, SEQ / 64), 128, 0, stream>>>(
      Zh, Woh, bo, out, SEQ, DMODEL, DMODEL);
}
